// PixelTimeLSTM_69896297775184
// MI455X (gfx1250) — compile-verified, matrix operand order checked by reading, not run on hardware
//
#include <hip/hip_runtime.h>
#include <hip/hip_bf16.h>

typedef __attribute__((ext_vector_type(16))) _Float16 v16h;
typedef __attribute__((ext_vector_type(8)))  _Float16 v8h;
typedef __attribute__((ext_vector_type(8)))  float    v8f;

constexpr int kB   = 8;
constexpr int kC   = 64;
constexpr int kHW  = 64 * 64;          // 4096
constexpr int kT   = 8;
constexpr int kG   = 8;
constexpr int kHID = 64;
constexpr int kN   = kB * kG * kHW;    // 262144 sequences
constexpr int kPitch = 72;             // padded LDS row pitch (halves): 144B, conflict-free b128

// ---------------------------------------------------------------------------
// helpers
// ---------------------------------------------------------------------------
__device__ __forceinline__ v16h ld_frag(const _Float16* p) {
  // two 16-byte LDS loads -> one 16-half WMMA operand fragment
  v8h lo = *(const v8h*)(p);
  v8h hh = *(const v8h*)(p + 16);
  return __builtin_shufflevector(lo, hh, 0,1,2,3,4,5,6,7,8,9,10,11,12,13,14,15);
}

// gfx1250 has V_TANH_F32 (transcendental, co-executes). Fall back to fast rcp.
__device__ __forceinline__ float fast_tanh(float x) {
#if __has_builtin(__builtin_amdgcn_tanhf)
  return __builtin_amdgcn_tanhf(x);
#else
  return fmaf(2.0f, __builtin_amdgcn_rcpf(1.0f + __expf(-2.0f * x)), -1.0f);
#endif
}
__device__ __forceinline__ float sigm(float x) {
  return fmaf(0.5f, fast_tanh(0.5f * x), 0.5f);   // sigmoid(x) = 0.5*tanh(x/2)+0.5
}
__device__ __forceinline__ void sched_fence() {
#if __has_builtin(__builtin_amdgcn_sched_barrier)
  __builtin_amdgcn_sched_barrier(0);              // cap cross-group code motion
#endif
}

// ---------------------------------------------------------------------------
// Stage 1: input 1x1 conv (C=64 -> G*T=64) + transpose into x[T][N]
// grid (kB*kHW/256, 4), block 256
// ---------------------------------------------------------------------------
__global__ __launch_bounds__(256) void in_conv_kernel(
    const float* __restrict__ hsrc, const float* __restrict__ w_in,
    const float* __restrict__ b_in, float* __restrict__ xbuf)
{
  const int pix = blockIdx.x * 256 + threadIdx.x;   // b*4096 + hw
  const int b  = pix >> 12;
  const int hw = pix & 4095;
  float hreg[kC];
#pragma unroll
  for (int c = 0; c < kC; ++c)
    hreg[c] = hsrc[(((long long)b * kC + c) << 12) + hw];
  const int o0 = blockIdx.y * 16;
#pragma unroll
  for (int oo = 0; oo < 16; ++oo) {
    const int o = o0 + oo;
    float acc = b_in[o];                            // uniform -> scalar load
    const float* wr = w_in + o * kC;
#pragma unroll
    for (int c = 0; c < kC; ++c) acc = fmaf(wr[c], hreg[c], acc);
    const int t = o & 7, g = o >> 3;
    xbuf[(long long)t * kN + (((long long)b * kG + g) << 12) + hw] = acc;
  }
}

// ---------------------------------------------------------------------------
// Stage 2: 262144 independent LSTMs, WMMA recurrent GEMM.
// One wave = 16 sequences. WG = 8 waves. grid kN/128 = 2048, block 256.
// gates(16x256) = h(16x64) @ w_hh^T via v_wmma_f32_16x16x32_f16,
// C seeded with bias + x_t * w_x. s_t = h_t @ w_s also via WMMA, computed
// from the A fragments loaded at iteration t+1 (h_t is next step's A input).
// ---------------------------------------------------------------------------
__global__ __launch_bounds__(256) void lstm_seq_kernel(
    const float* __restrict__ xbuf, const float* __restrict__ whh,
    const float* __restrict__ wih,  const float* __restrict__ bih,
    const float* __restrict__ bhh,  const float* __restrict__ wsv,
    const float* __restrict__ bsv,  float* __restrict__ sbuf)
{
  __shared__ _Float16 sWhh[256 * kPitch];   // w_hh in f16, row-major, padded pitch
  __shared__ _Float16 sH[8][16 * kPitch];   // per-wave h-state tile (16 x 64 used)
  __shared__ float sBias[256];
  __shared__ float sWx[256];
  __shared__ float sWs[kHID];

  const int tid = threadIdx.x;
  {
    const float* src = whh + tid * kHID;    // thread tid owns row tid of w_hh
    _Float16* dst = sWhh + tid * kPitch;
#pragma unroll
    for (int c = 0; c < kHID; ++c) dst[c] = (_Float16)src[c];
    sBias[tid] = bih[tid] + bhh[tid];
    sWx[tid]   = wih[tid];                  // w_ih[:,0]
    if (tid < kHID) sWs[tid] = wsv[tid];
  }
  const int wave = tid >> 5;
  const int lane = tid & 31;
  const int j16  = lane & 15;
  const int hi   = lane >> 4;               // 0: lanes 0-15, 1: lanes 16-31
  _Float16* hT = sH[wave];
  for (int i = lane; i < 16 * kPitch; i += 32) hT[i] = (_Float16)0.0f;
  __syncthreads();

  // constant B fragments holding w_s in output column 0 only (lanes j16==0)
  v16h bs0, bs1;
#pragma unroll
  for (int i = 0; i < 16; ++i) { bs0[i] = (_Float16)0.0f; bs1[i] = (_Float16)0.0f; }
  if (j16 == 0) {
#pragma unroll
    for (int i = 0; i < 8; ++i) {
      bs0[i]     = (_Float16)sWs[ 0 + hi * 8 + i];
      bs0[8 + i] = (_Float16)sWs[16 + hi * 8 + i];
      bs1[i]     = (_Float16)sWs[32 + hi * 8 + i];
      bs1[8 + i] = (_Float16)sWs[48 + hi * 8 + i];
    }
  }

  const long long n0 = (long long)blockIdx.x * 128 + wave * 16;
  const float bS = bsv[0];

  v8f cst[4];                               // cell state, 4 col-chunks of 16
#pragma unroll
  for (int nc = 0; nc < 4; ++nc)
#pragma unroll
    for (int v = 0; v < 8; ++v) cst[nc][v] = 0.0f;

  for (int t = 0; t < kT; ++t) {
    // A fragments: h-state, K chunks 0..31 / 32..63 (wave32 16-bit A layout)
    v16h a0 = ld_frag(hT + j16 * kPitch + 0  + hi * 8);
    v16h a1 = ld_frag(hT + j16 * kPitch + 32 + hi * 8);

    // s_{t-1} = h_{t-1} @ w_s via WMMA on the freshly loaded A fragments.
    if (t > 0) {
      v8f sv;
#pragma unroll
      for (int v = 0; v < 8; ++v) sv[v] = 0.0f;
      sv = __builtin_amdgcn_wmma_f32_16x16x32_f16(false, a0, false, bs0,
                                                  (short)0, sv, false, false);
      sv = __builtin_amdgcn_wmma_f32_16x16x32_f16(false, a1, false, bs1,
                                                  (short)0, sv, false, false);
      if (j16 == 0) {                       // column 0 lives on lanes 0 and 16
#pragma unroll
        for (int v = 0; v < 8; ++v)
          sbuf[(long long)(t - 1) * kN + n0 + v + hi * 8] = sv[v] + bS;
      }
    }

    // per-half broadcast vector loads: xm[v] = x_t[row v + 8*hi]
    const float* xrow = xbuf + (long long)t * kN + n0 + hi * 8;
    const float4 xv0 = *(const float4*)(xrow);
    const float4 xv1 = *(const float4*)(xrow + 4);
    float xm[8];
    xm[0] = xv0.x; xm[1] = xv0.y; xm[2] = xv0.z; xm[3] = xv0.w;
    xm[4] = xv1.x; xm[5] = xv1.y; xm[6] = xv1.z; xm[7] = xv1.w;

#pragma unroll
    for (int nc = 0; nc < 4; ++nc) {        // hidden columns 16*nc .. 16*nc+15
      v8f g4[4];                            // i, f, g, o fragments
#pragma unroll
      for (int q = 0; q < 4; ++q) {
        const int jc  = nc + q * 4;         // gate output block (of 16)
        const int col = jc * 16 + j16;
        const float bb = sBias[col];
        const float wx = sWx[col];
        v8f acc;
#pragma unroll
        for (int v = 0; v < 8; ++v) acc[v] = fmaf(xm[v], wx, bb);
        v16h b0 = ld_frag(sWhh + col * kPitch + 0  + hi * 8);
        acc = __builtin_amdgcn_wmma_f32_16x16x32_f16(false, a0, false, b0,
                                                     (short)0, acc, false, false);
        v16h b1 = ld_frag(sWhh + col * kPitch + 32 + hi * 8);
        acc = __builtin_amdgcn_wmma_f32_16x16x32_f16(false, a1, false, b1,
                                                     (short)0, acc, false, false);
        g4[q] = acc;
      }
#pragma unroll
      for (int v = 0; v < 8; ++v) {         // row m = v + 8*hi, col = nc*16 + j16
        float iv = sigm(g4[0][v]);
        float fv = sigm(g4[1][v]);
        float gv = fast_tanh(g4[2][v]);
        float ov = sigm(g4[3][v]);
        float cn = fmaf(fv, cst[nc][v], iv * gv);
        cst[nc][v] = cn;
        float hn = ov * fast_tanh(cn);
        hT[(v + hi * 8) * kPitch + nc * 16 + j16] = (_Float16)hn;
      }
      sched_fence();                        // keep only one nc group in flight
    }
  }

  // epilogue: s for the final step (h_{T-1} is in the LDS tile now)
  {
    v16h a0 = ld_frag(hT + j16 * kPitch + 0  + hi * 8);
    v16h a1 = ld_frag(hT + j16 * kPitch + 32 + hi * 8);
    v8f sv;
#pragma unroll
    for (int v = 0; v < 8; ++v) sv[v] = 0.0f;
    sv = __builtin_amdgcn_wmma_f32_16x16x32_f16(false, a0, false, bs0,
                                                (short)0, sv, false, false);
    sv = __builtin_amdgcn_wmma_f32_16x16x32_f16(false, a1, false, bs1,
                                                (short)0, sv, false, false);
    if (j16 == 0) {
#pragma unroll
      for (int v = 0; v < 8; ++v)
        sbuf[(long long)(kT - 1) * kN + n0 + v + hi * 8] = sv[v] + bS;
    }
  }
}

// ---------------------------------------------------------------------------
// Stage 3: gather s[T][N] back per pixel and apply output 1x1 conv (64 -> 64)
// grid (kB*kHW/256, 4), block 256
// ---------------------------------------------------------------------------
__global__ __launch_bounds__(256) void out_conv_kernel(
    const float* __restrict__ sbuf, const float* __restrict__ w_out,
    const float* __restrict__ b_out, float* __restrict__ out)
{
  const int pix = blockIdx.x * 256 + threadIdx.x;
  const int b  = pix >> 12;
  const int hw = pix & 4095;
  float sreg[64];
#pragma unroll
  for (int g = 0; g < kG; ++g)
#pragma unroll
    for (int t = 0; t < kT; ++t)
      sreg[g * kT + t] =
          sbuf[(long long)t * kN + (((long long)b * kG + g) << 12) + hw];
  const int oc0 = blockIdx.y * 16;
#pragma unroll
  for (int k = 0; k < 16; ++k) {
    const int oc = oc0 + k;
    float acc = b_out[oc];
    const float* wr = w_out + oc * 64;
#pragma unroll
    for (int j = 0; j < 64; ++j) acc = fmaf(wr[j], sreg[j], acc);
    out[(((long long)b * kC + oc) << 12) + hw] = acc;
  }
}

// ---------------------------------------------------------------------------
extern "C" void kernel_launch(void* const* d_in, const int* in_sizes, int n_in,
                              void* d_out, int out_size, void* d_ws, size_t ws_size,
                              hipStream_t stream) {
  const float* h     = (const float*)d_in[0];
  const float* w_in  = (const float*)d_in[1];
  const float* b_in  = (const float*)d_in[2];
  const float* w_ih  = (const float*)d_in[3];
  const float* w_hh  = (const float*)d_in[4];
  const float* b_ih  = (const float*)d_in[5];
  const float* b_hh  = (const float*)d_in[6];
  const float* w_s   = (const float*)d_in[7];
  const float* b_s   = (const float*)d_in[8];
  const float* w_out = (const float*)d_in[9];
  const float* b_out = (const float*)d_in[10];
  float* out = (float*)d_out;

  float* xbuf = (float*)d_ws;                       // [T][N] fp32, 8 MB
  float* sbuf = xbuf + (size_t)kT * kN;             // [T][N] fp32, 8 MB

  in_conv_kernel<<<dim3((kB * kHW) / 256, 4), 256, 0, stream>>>(h, w_in, b_in, xbuf);
  lstm_seq_kernel<<<kN / 128, 256, 0, stream>>>(xbuf, w_hh, w_ih, b_ih, b_hh,
                                                w_s, b_s, sbuf);
  out_conv_kernel<<<dim3((kB * kHW) / 256, 4), 256, 0, stream>>>(sbuf, w_out, b_out, out);
}
